// Encoder_49873160241164
// MI455X (gfx1250) — matrix operand order repaired and checked by reading, not compiled, not run on hardware
//
#include <hip/hip_runtime.h>
#include <hip/hip_bf16.h>
#include <hip/hip_fp16.h>
#include <math.h>

typedef __attribute__((ext_vector_type(16))) _Float16 v16h;
typedef __attribute__((ext_vector_type(8)))  _Float16 v8h;
typedef __attribute__((ext_vector_type(8)))  float    v8f;

#define BB   64
#define PP   96
#define NWIN 12
#define DD   2048
#define WW   85
#define SCALE_QK 0.022097086912079608f   // 1/sqrt(2048)
#define LN_EPS 1e-5f

union HFrag { v16h v; v8h h[2]; };

// ---------------------------------------------------------------------------
// f32 -> f16 conversion (grid-stride)
// ---------------------------------------------------------------------------
__global__ __launch_bounds__(256)
void cvt_f16_kernel(const float* __restrict__ s, _Float16* __restrict__ d, long n) {
  long i = (long)blockIdx.x * blockDim.x + threadIdx.x;
  long stride = (long)gridDim.x * blockDim.x;
  for (; i < n; i += stride) d[i] = (_Float16)s[i];
}

// ---------------------------------------------------------------------------
// Generic batched WMMA GEMM: C[b] = A[b] * B[b]^T (+bias), A:[M,K], B:[N,K]
// row-major f16, C row-major (f32 or f16). 32x32 wave tile.
// ---------------------------------------------------------------------------
__global__ __launch_bounds__(256)
void gemm_wmma_kernel(const _Float16* __restrict__ A, const _Float16* __restrict__ Bw,
                      const float* __restrict__ bias,
                      float* __restrict__ Cf, _Float16* __restrict__ Ch,
                      int M, int N, int K,
                      long sA, long sB, long sC,
                      int tilesM, int tilesN, int nTiles) {
  int wave = blockIdx.x * (blockDim.x >> 5) + (threadIdx.x >> 5);
  if (wave >= nTiles) return;                 // uniform per-wave branch
  int lane   = threadIdx.x & 31;
  int lane16 = lane & 15;
  int half   = lane >> 4;

  int perB = tilesM * tilesN;
  int b    = wave / perB;
  int t    = wave % perB;
  int m0   = (t / tilesN) * 32;
  int n0   = (t % tilesN) * 32;

  const _Float16* Ab = A  + (long)b * sA;
  const _Float16* Bb = Bw + (long)b * sB;

  // A frag row = m0 + lane16 ; K halves: [kb+half*16 .. +15]
  const _Float16* a0p = Ab + (size_t)(m0 + lane16) * K + half * 16;
  const _Float16* a1p = a0p + (size_t)16 * K;
  // B frag col = n0 + lane16 ; K halves: [kb+half*16 .. +15]
  const _Float16* b0p = Bb + (size_t)(n0 + lane16) * K + half * 16;
  const _Float16* b1p = b0p + (size_t)16 * K;

  v8f c00 = {}, c01 = {}, c10 = {}, c11 = {};

  for (int kb = 0; kb < K; kb += 32) {
    HFrag a0, a1, b0, b1;
    a0.h[0] = *(const v8h*)(a0p + kb);
    a0.h[1] = *(const v8h*)(a0p + kb + 8);
    a1.h[0] = *(const v8h*)(a1p + kb);
    a1.h[1] = *(const v8h*)(a1p + kb + 8);
    b0.h[0] = *(const v8h*)(b0p + kb);
    b0.h[1] = *(const v8h*)(b0p + kb + 8);
    b1.h[0] = *(const v8h*)(b1p + kb);
    b1.h[1] = *(const v8h*)(b1p + kb + 8);
    c00 = __builtin_amdgcn_wmma_f32_16x16x32_f16(false, a0.v, false, b0.v, (short)0, c00, false, false);
    c01 = __builtin_amdgcn_wmma_f32_16x16x32_f16(false, a0.v, false, b1.v, (short)0, c01, false, false);
    c10 = __builtin_amdgcn_wmma_f32_16x16x32_f16(false, a1.v, false, b0.v, (short)0, c10, false, false);
    c11 = __builtin_amdgcn_wmma_f32_16x16x32_f16(false, a1.v, false, b1.v, (short)0, c11, false, false);
  }

  int col0 = n0 + lane16, col1 = col0 + 16;
  float bv0 = bias ? bias[col0] : 0.0f;
  float bv1 = bias ? bias[col1] : 0.0f;
  long base = (long)b * sC;
  if (Ch) {
#pragma unroll
    for (int i = 0; i < 8; ++i) {
      int r0 = m0 + half * 8 + i, r1 = r0 + 16;
      Ch[base + (long)r0 * N + col0] = (_Float16)(c00[i] + bv0);
      Ch[base + (long)r0 * N + col1] = (_Float16)(c01[i] + bv1);
      Ch[base + (long)r1 * N + col0] = (_Float16)(c10[i] + bv0);
      Ch[base + (long)r1 * N + col1] = (_Float16)(c11[i] + bv1);
    }
  } else {
#pragma unroll
    for (int i = 0; i < 8; ++i) {
      int r0 = m0 + half * 8 + i, r1 = r0 + 16;
      Cf[base + (long)r0 * N + col0] = c00[i] + bv0;
      Cf[base + (long)r0 * N + col1] = c01[i] + bv1;
      Cf[base + (long)r1 * N + col0] = c10[i] + bv0;
      Cf[base + (long)r1 * N + col1] = c11[i] + bv1;
    }
  }
}

// ---------------------------------------------------------------------------
// High-reuse WMMA GEMM for the big D x D weight GEMMs (batch=1, N % 64 == 0):
// 32x64 wave tile -> each A fragment feeds 4 B fragments (8 WMMAs / K-step).
// ---------------------------------------------------------------------------
__global__ __launch_bounds__(256)
void gemm_wmma64_kernel(const _Float16* __restrict__ A, const _Float16* __restrict__ Bw,
                        const float* __restrict__ bias,
                        float* __restrict__ Cf, _Float16* __restrict__ Ch,
                        int N, int K, int tilesN, int nTiles) {
  int wave = blockIdx.x * (blockDim.x >> 5) + (threadIdx.x >> 5);
  if (wave >= nTiles) return;                 // uniform per-wave branch
  int lane   = threadIdx.x & 31;
  int lane16 = lane & 15;
  int half   = lane >> 4;

  int m0 = (wave / tilesN) * 32;
  int n0 = (wave % tilesN) * 64;

  const _Float16* a0p = A + (size_t)(m0 + lane16) * K + half * 16;
  const _Float16* a1p = a0p + (size_t)16 * K;
  const _Float16* bp0 = Bw + (size_t)(n0 + lane16) * K + half * 16;
  const _Float16* bp1 = bp0 + (size_t)16 * K;
  const _Float16* bp2 = bp0 + (size_t)32 * K;
  const _Float16* bp3 = bp0 + (size_t)48 * K;

  v8f c00 = {}, c01 = {}, c02 = {}, c03 = {};
  v8f c10 = {}, c11 = {}, c12 = {}, c13 = {};

  for (int kb = 0; kb < K; kb += 32) {
    HFrag a0, a1, b0, b1, b2, b3;
    a0.h[0] = *(const v8h*)(a0p + kb);
    a0.h[1] = *(const v8h*)(a0p + kb + 8);
    a1.h[0] = *(const v8h*)(a1p + kb);
    a1.h[1] = *(const v8h*)(a1p + kb + 8);
    b0.h[0] = *(const v8h*)(bp0 + kb);
    b0.h[1] = *(const v8h*)(bp0 + kb + 8);
    b1.h[0] = *(const v8h*)(bp1 + kb);
    b1.h[1] = *(const v8h*)(bp1 + kb + 8);
    b2.h[0] = *(const v8h*)(bp2 + kb);
    b2.h[1] = *(const v8h*)(bp2 + kb + 8);
    b3.h[0] = *(const v8h*)(bp3 + kb);
    b3.h[1] = *(const v8h*)(bp3 + kb + 8);
    c00 = __builtin_amdgcn_wmma_f32_16x16x32_f16(false, a0.v, false, b0.v, (short)0, c00, false, false);
    c10 = __builtin_amdgcn_wmma_f32_16x16x32_f16(false, a1.v, false, b0.v, (short)0, c10, false, false);
    c01 = __builtin_amdgcn_wmma_f32_16x16x32_f16(false, a0.v, false, b1.v, (short)0, c01, false, false);
    c11 = __builtin_amdgcn_wmma_f32_16x16x32_f16(false, a1.v, false, b1.v, (short)0, c11, false, false);
    c02 = __builtin_amdgcn_wmma_f32_16x16x32_f16(false, a0.v, false, b2.v, (short)0, c02, false, false);
    c12 = __builtin_amdgcn_wmma_f32_16x16x32_f16(false, a1.v, false, b2.v, (short)0, c12, false, false);
    c03 = __builtin_amdgcn_wmma_f32_16x16x32_f16(false, a0.v, false, b3.v, (short)0, c03, false, false);
    c13 = __builtin_amdgcn_wmma_f32_16x16x32_f16(false, a1.v, false, b3.v, (short)0, c13, false, false);
  }

  v8f* c0a[4] = { &c00, &c01, &c02, &c03 };
  v8f* c1a[4] = { &c10, &c11, &c12, &c13 };
#pragma unroll
  for (int j = 0; j < 4; ++j) {
    int col = n0 + j * 16 + lane16;
    float bv = bias ? bias[col] : 0.0f;
    v8f v0 = *c0a[j], v1 = *c1a[j];
    if (Ch) {
#pragma unroll
      for (int i = 0; i < 8; ++i) {
        int r0 = m0 + half * 8 + i;
        Ch[(long)r0 * N + col]        = (_Float16)(v0[i] + bv);
        Ch[(long)(r0 + 16) * N + col] = (_Float16)(v1[i] + bv);
      }
    } else {
#pragma unroll
      for (int i = 0; i < 8; ++i) {
        int r0 = m0 + half * 8 + i;
        Cf[(long)r0 * N + col]        = v0[i] + bv;
        Cf[(long)(r0 + 16) * N + col] = v1[i] + bv;
      }
    }
  }
}

// ---------------------------------------------------------------------------
// Batch-independent window masks: wmask[12] -> masks[0..11], wmask2[85] -> masks[16..100]
// ---------------------------------------------------------------------------
__global__ void masks_kernel(const float* __restrict__ nfw, const float* __restrict__ nfb,
                             const float* __restrict__ sfw, const float* __restrict__ sfb,
                             float* __restrict__ masks) {
  __shared__ float t1[NWIN];
  __shared__ float t2[WW];
  int tid = threadIdx.x;
  if (tid < NWIN) { float s = nfb[tid]; for (int n = 0; n < NWIN; ++n) s += nfw[tid * NWIN + n]; t1[tid] = s; }
  if (tid < WW)   { float s = sfb[tid]; for (int u = 0; u < WW;   ++u) s += sfw[tid * WW + u];   t2[tid] = s; }
  __syncthreads();
  if (tid == 0) {
    float mx = -3.4e38f; for (int j = 0; j < NWIN; ++j) mx = fmaxf(mx, t1[j]);
    float sm = 0.f; float e[NWIN];
    for (int j = 0; j < NWIN; ++j) { e[j] = expf(t1[j] - mx); sm += e[j]; }
    for (int j = 0; j < NWIN; ++j) masks[j] = e[j] / sm;
  }
  if (tid == 1) {
    float mx = -3.4e38f; for (int j = 0; j < WW; ++j) mx = fmaxf(mx, t2[j]);
    float sm = 0.f;
    for (int j = 0; j < WW; ++j) { t2[j] = expf(t2[j] - mx); sm += t2[j]; }
    for (int j = 0; j < WW; ++j) masks[16 + j] = t2[j] / sm;
  }
}

// ---------------------------------------------------------------------------
// Window attention coefficients: c[b,w,m] = sum_n wmask[n]*softmax_m(S1[b,w+n,w+m]*SCALE)
// ---------------------------------------------------------------------------
__global__ __launch_bounds__(256)
void cwin_kernel(const float* __restrict__ S1, const float* __restrict__ masks,
                 float* __restrict__ cbuf) {
  int idx = blockIdx.x * blockDim.x + threadIdx.x;
  if (idx >= BB * WW) return;
  int b = idx / WW, w = idx % WW;
  const float* Sb = S1 + (long)b * PP * PP;
  float cc[NWIN];
  for (int m = 0; m < NWIN; ++m) cc[m] = 0.f;
  for (int n = 0; n < NWIN; ++n) {
    float row[NWIN]; float mx = -3.4e38f;
    for (int m = 0; m < NWIN; ++m) {
      row[m] = Sb[(w + n) * PP + (w + m)] * SCALE_QK;
      mx = fmaxf(mx, row[m]);
    }
    float sm = 0.f;
    for (int m = 0; m < NWIN; ++m) { row[m] = expf(row[m] - mx); sm += row[m]; }
    float wn = masks[n] / sm;
    for (int m = 0; m < NWIN; ++m) cc[m] += wn * row[m];
  }
  for (int m = 0; m < NWIN; ++m) cbuf[(long)idx * NWIN + m] = cc[m];
}

// ---------------------------------------------------------------------------
// x[b,w,d] = sum_p ds_w[w,p]*fc[b,p,d] + ds_b[w] + sum_m c[b,w,m]*vp[b,w+m,d]
// ---------------------------------------------------------------------------
__global__ __launch_bounds__(256)
void build_x_kernel(const float* __restrict__ fc, const float* __restrict__ vp,
                    const float* __restrict__ cbuf, const float* __restrict__ dsw,
                    const float* __restrict__ dsb, float* __restrict__ x) {
  const int chunks = DD / 256;
  int d  = threadIdx.x + (blockIdx.x % chunks) * 256;
  int bw = blockIdx.x / chunks;
  int b = bw / WW, w = bw % WW;
  const float* fcb = fc + (long)b * PP * DD + d;
  float acc = dsb[w];
  const float* dw = dsw + w * PP;
  for (int p = 0; p < PP; ++p) acc += dw[p] * fcb[(long)p * DD];
  const float* vb = vp + ((long)b * PP + w) * DD + d;
  const float* cw = cbuf + (long)bw * NWIN;
  for (int m = 0; m < NWIN; ++m) acc += cw[m] * vb[(long)m * DD];
  x[((long)b * PP + w) * DD + d] = acc;
}

// ---------------------------------------------------------------------------
// LayerNorm over d (in place, f32) + f16 copy; rows w>=85 zero-padded in xh.
// One block per (b, w<96).
// ---------------------------------------------------------------------------
__global__ __launch_bounds__(256)
void ln_kernel(float* __restrict__ x, _Float16* __restrict__ xh,
               const float* __restrict__ gam, const float* __restrict__ bet) {
  int bw = blockIdx.x;            // B*96
  int w = bw % PP;
  long base = (long)bw * DD;
  if (w >= WW) {
    for (int d = threadIdx.x; d < DD; d += 256) xh[base + d] = (_Float16)0.0f;
    return;
  }
  __shared__ float red[256];
  float loc[DD / 256];
  float s = 0.f;
#pragma unroll
  for (int i = 0; i < DD / 256; ++i) { loc[i] = x[base + threadIdx.x + i * 256]; s += loc[i]; }
  red[threadIdx.x] = s; __syncthreads();
  for (int st = 128; st > 0; st >>= 1) { if (threadIdx.x < st) red[threadIdx.x] += red[threadIdx.x + st]; __syncthreads(); }
  float mu = red[0] / (float)DD;
  __syncthreads();
  float v = 0.f;
#pragma unroll
  for (int i = 0; i < DD / 256; ++i) { float dv = loc[i] - mu; v += dv * dv; }
  red[threadIdx.x] = v; __syncthreads();
  for (int st = 128; st > 0; st >>= 1) { if (threadIdx.x < st) red[threadIdx.x] += red[threadIdx.x + st]; __syncthreads(); }
  float rstd = rsqrtf(red[0] / (float)DD + LN_EPS);
#pragma unroll
  for (int i = 0; i < DD / 256; ++i) {
    int d = threadIdx.x + i * 256;
    float y = (loc[i] - mu) * rstd * gam[d] + bet[d];
    x[base + d]  = y;
    xh[base + d] = (_Float16)y;
  }
}

// ---------------------------------------------------------------------------
// Row max / sum-exp of stage-2 scores (per (b, w<85), over u<85)
// ---------------------------------------------------------------------------
__global__ __launch_bounds__(256)
void rowstats_kernel(const float* __restrict__ S2, float* __restrict__ rs) {
  int idx = blockIdx.x * blockDim.x + threadIdx.x;
  if (idx >= BB * WW) return;
  int b = idx / WW, w = idx % WW;
  const float* r = S2 + ((long)b * PP + w) * PP;
  float mx = -3.4e38f;
  for (int u = 0; u < WW; ++u) mx = fmaxf(mx, r[u] * SCALE_QK);
  float sm = 0.f;
  for (int u = 0; u < WW; ++u) sm += expf(r[u] * SCALE_QK - mx);
  rs[idx * 2] = mx; rs[idx * 2 + 1] = sm;
}

// ---------------------------------------------------------------------------
// g[b,u] = sum_w wmask2[w] * attn2[b,w,u]   (deterministic, no atomics)
// ---------------------------------------------------------------------------
__global__ __launch_bounds__(256)
void g_kernel(const float* __restrict__ S2, const float* __restrict__ rs,
              const float* __restrict__ masks, float* __restrict__ gbuf) {
  int idx = blockIdx.x * blockDim.x + threadIdx.x;
  if (idx >= BB * WW) return;
  int b = idx / WW, u = idx % WW;
  const float* Sb = S2 + (long)b * PP * PP;
  const float* rb = rs + (long)b * WW * 2;
  const float* wm2 = masks + 16;
  float acc = 0.f;
  for (int w = 0; w < WW; ++w) {
    float e = expf(Sb[w * PP + u] * SCALE_QK - rb[w * 2]) / rb[w * 2 + 1];
    acc += wm2[w] * e;
  }
  gbuf[idx] = acc;
}

// ---------------------------------------------------------------------------
// y[b,d] = sum_u g[b,u] * x_ln[b,u,d]  -> f16 for final WMMA GEMM
// ---------------------------------------------------------------------------
__global__ __launch_bounds__(256)
void y_kernel(const float* __restrict__ x, const float* __restrict__ gbuf,
              _Float16* __restrict__ yh) {
  const int chunks = DD / 256;
  int d = threadIdx.x + (blockIdx.x % chunks) * 256;
  int b = blockIdx.x / chunks;
  const float* xb = x + (long)b * PP * DD + d;
  const float* gb = gbuf + b * WW;
  float acc = 0.f;
  for (int u = 0; u < WW; ++u) acc += gb[u] * xb[(long)u * DD];
  yh[(long)b * DD + d] = (_Float16)acc;
}

// ---------------------------------------------------------------------------
extern "C" void kernel_launch(void* const* d_in, const int* in_sizes, int n_in,
                              void* d_out, int out_size, void* d_ws, size_t ws_size,
                              hipStream_t stream) {
  const float* fc  = (const float*)d_in[0];
  const float* qw1 = (const float*)d_in[1];
  const float* qb1 = (const float*)d_in[2];
  const float* kw1 = (const float*)d_in[3];
  const float* kb1 = (const float*)d_in[4];
  const float* vw1 = (const float*)d_in[5];
  const float* vb1 = (const float*)d_in[6];
  const float* nfw = (const float*)d_in[7];
  const float* nfb = (const float*)d_in[8];
  const float* dsw = (const float*)d_in[9];
  const float* dsb = (const float*)d_in[10];
  const float* lng = (const float*)d_in[11];
  const float* lnb = (const float*)d_in[12];
  const float* qw2 = (const float*)d_in[13];
  const float* qb2 = (const float*)d_in[14];
  const float* kw2 = (const float*)d_in[15];
  const float* kb2 = (const float*)d_in[16];
  const float* vw2 = (const float*)d_in[17];
  const float* vb2 = (const float*)d_in[18];
  const float* sfw = (const float*)d_in[19];
  const float* sfb = (const float*)d_in[20];
  float* out = (float*)d_out;
  (void)in_sizes; (void)n_in; (void)out_size; (void)ws_size;

  char* ws = (char*)d_ws;
  size_t off = 0;
  auto alloc = [&](size_t bytes) -> char* {
    char* p = ws + off; off += (bytes + 255) & ~(size_t)255; return p;
  };
  const long TOK  = (long)BB * PP;          // 6144 padded tokens
  const long TOKD = TOK * DD;               // 12,582,912
  _Float16* fcH  = (_Float16*)alloc(TOKD * 2);
  _Float16* wH[6];
  for (int i = 0; i < 6; ++i) wH[i] = (_Float16*)alloc((size_t)DD * DD * 2);
  _Float16* qpH  = (_Float16*)alloc(TOKD * 2);   // later reused for q2
  _Float16* kpH  = (_Float16*)alloc(TOKD * 2);   // later reused for k2
  float*    vp   = (float*)   alloc(TOKD * 4);
  float*    S    = (float*)   alloc((size_t)BB * PP * PP * 4);  // S1 then S2
  float*    cb   = (float*)   alloc((size_t)BB * WW * NWIN * 4);
  float*    xb   = (float*)   alloc(TOKD * 4);
  _Float16* xH   = (_Float16*)alloc(TOKD * 2);
  float*    masks= (float*)   alloc(128 * 4);
  float*    rs   = (float*)   alloc((size_t)BB * WW * 2 * 4);
  float*    gb   = (float*)   alloc((size_t)BB * WW * 4);
  _Float16* yH   = (_Float16*)alloc((size_t)BB * DD * 2);

  const float* wsrc[6] = { qw1, kw1, vw1, qw2, kw2, vw2 };

  // 1) conversions
  cvt_f16_kernel<<<4096, 256, 0, stream>>>(fc, fcH, TOKD);
  for (int i = 0; i < 6; ++i)
    cvt_f16_kernel<<<4096, 256, 0, stream>>>(wsrc[i], wH[i], (long)DD * DD);

  // 2) constant masks
  masks_kernel<<<1, 128, 0, stream>>>(nfw, nfb, sfw, sfb, masks);

  // big GEMMs (batch=1, N%64==0) use the 32x64 high-reuse kernel
  auto launch_gemm_big = [&](const _Float16* A, const _Float16* Bw, const float* bias,
                             float* Cf, _Float16* Ch, int M, int N, int K) {
    int tN = N / 64;
    int nTiles = (M / 32) * tN;
    int blocks = (nTiles + 7) / 8;
    gemm_wmma64_kernel<<<blocks, 256, 0, stream>>>(A, Bw, bias, Cf, Ch, N, K, tN, nTiles);
  };
  // batched / small-N GEMMs use the generic 32x32 kernel
  auto launch_gemm = [&](const _Float16* A, const _Float16* Bw, const float* bias,
                         float* Cf, _Float16* Ch, int M, int N, int K,
                         long sA, long sB, long sC, int nBatch) {
    int tM = M / 32, tN = N / 32;
    int nTiles = tM * tN * nBatch;
    int blocks = (nTiles + 7) / 8;
    gemm_wmma_kernel<<<blocks, 256, 0, stream>>>(A, Bw, bias, Cf, Ch,
                                                 M, N, K, sA, sB, sC, tM, tN, nTiles);
  };

  // 3) stage-1 projections on all 6144 rows (window trick)
  launch_gemm_big(fcH, wH[0], qb1, nullptr, qpH, PP * BB, DD, DD);
  launch_gemm_big(fcH, wH[1], kb1, nullptr, kpH, PP * BB, DD, DD);
  launch_gemm_big(fcH, wH[2], vb1, vp, nullptr, PP * BB, DD, DD);

  // 4) stage-1 scores S1[b] = qp[b] * kp[b]^T  (96x96 per batch)
  launch_gemm(qpH, kpH, nullptr, S, nullptr, PP, PP, DD,
              (long)PP * DD, (long)PP * DD, (long)PP * PP, BB);

  // 5) banded softmax -> combine coefficients c[b,w,m]
  cwin_kernel<<<(BB * WW + 255) / 256, 256, 0, stream>>>(S, masks, cb);

  // 6) x = downsample(fc) + windowed-attn(vp)
  build_x_kernel<<<BB * WW * (DD / 256), 256, 0, stream>>>(fc, vp, cb, dsw, dsb, xb);

  // 7) LayerNorm (in-place f32 + f16 copy, zero-padded to 96 rows/batch)
  ln_kernel<<<BB * PP, 256, 0, stream>>>(xb, xH, lng, lnb);

  // 8) stage-2 q/k projections (reuse qpH/kpH buffers)
  launch_gemm_big(xH, wH[3], qb2, nullptr, qpH, PP * BB, DD, DD);
  launch_gemm_big(xH, wH[4], kb2, nullptr, kpH, PP * BB, DD, DD);

  // 9) stage-2 scores S2[b] = q2[b] * k2[b]^T (reuse S buffer)
  launch_gemm(qpH, kpH, nullptr, S, nullptr, PP, PP, DD,
              (long)PP * DD, (long)PP * DD, (long)PP * PP, BB);

  // 10) softmax stats, attention-collapse weights g, y = sum_u g*x_ln
  rowstats_kernel<<<(BB * WW + 255) / 256, 256, 0, stream>>>(S, rs);
  g_kernel<<<(BB * WW + 255) / 256, 256, 0, stream>>>(S, rs, masks, gb);
  y_kernel<<<BB * (DD / 256), 256, 0, stream>>>(xb, gb, yH);

  // 11) out = y @ sa_v_w^T + sa_v_b   (64x2048, bias applies once since sum(g)=1)
  launch_gemm_big(yH, wH[5], vb2, out, nullptr, BB, DD, DD);
}
